// MRSFFBlock_88854283419799
// MI455X (gfx1250) — matrix operand order repaired and checked by reading, not compiled, not run on hardware
//
#include <hip/hip_runtime.h>

// ---------------------------------------------------------------------------
// MRSFF block for gfx1250 (MI455X). All GEMM-like math runs on
// v_wmma_f32_16x16x32_f16 (wave32 WMMA, f32 accumulate). Activations and
// weights are kept in f16 to halve HBM traffic (23.3 TB/s roofline); all
// pointwise math (LN/softmax/GELU) stays f32.
// ---------------------------------------------------------------------------

typedef _Float16 f16;
typedef _Float16 v16h __attribute__((ext_vector_type(16)));
typedef _Float16 v8h  __attribute__((ext_vector_type(8)));
typedef float    v8f  __attribute__((ext_vector_type(8)));

#define DIMC 256
#define NH   8
#define HD   32
#define NTOK 64          // window tokens (8x8)
#define TT   4
#define MM   5
#define BB   2
#define NW   64          // windows per image (8x8 of 8x8)
#define SCALEQ 0.17677669529663687f

__device__ __forceinline__ v8f wmma_f16(v16h a, v16h b, v8f c) {
  return __builtin_amdgcn_wmma_f32_16x16x32_f16(false, a, false, b, (short)0, c, false, false);
}

// A-fragment: lane holds row (lane&15); K = kb..kb+7 in v0..3, kb+16..kb+23 in
// v4..7 where kb = (lane>>4)*8. Caller passes p = &row[kb].
__device__ __forceinline__ v16h ldfragA(const f16* p) {
  v8h lo = *(const v8h*)p;
  v8h hi = *(const v8h*)(p + 16);
  v16h a;
#pragma unroll
  for (int i = 0; i < 8; ++i) { a[i] = lo[i]; a[i + 8] = hi[i]; }
  return a;
}

// B-fragment: lane holds column (lane&15); K = kb..kb+7 in v0..3, kb+16..kb+23 in
// v4..7 where kb = (lane>>4)*8. Caller passes p = &col[kb] (K contiguous in memory).
__device__ __forceinline__ v16h ldfragB(const f16* p) {
  v8h lo = *(const v8h*)p;
  v8h hi = *(const v8h*)(p + 16);
  v16h b;
#pragma unroll
  for (int i = 0; i < 8; ++i) { b[i] = lo[i]; b[i + 8] = hi[i]; }
  return b;
}

__device__ __forceinline__ float warp_sum(float v) {
#pragma unroll
  for (int o = 16; o > 0; o >>= 1) v += __shfl_xor(v, o, 32);
  return v;
}

__device__ __forceinline__ float gelu_exact(float x) {
  return 0.5f * x * (1.0f + erff(x * 0.70710678118654752f));
}

// ---------------------------------------------------------------------------
// One-time f32 -> f16 weight conversion (weights then live in L2: ~2 MB).
// ---------------------------------------------------------------------------
__global__ void cvt_w_kernel(const float* __restrict__ src, f16* __restrict__ dst, int n) {
  int i = (blockIdx.x * 256 + threadIdx.x) * 4;
  if (i + 3 < n) {
    float4 v = *(const float4*)(src + i);
    dst[i + 0] = (f16)v.x; dst[i + 1] = (f16)v.y;
    dst[i + 2] = (f16)v.z; dst[i + 3] = (f16)v.w;
  } else {
    for (int j = i; j < n; ++j) dst[j] = (f16)src[j];
  }
}

// ---------------------------------------------------------------------------
// LayerNorm (f32 in) -> window-partitioned f16 rows ((b_*S+s)*64+n, C)
// One wave per token, lane handles 8 channels.
// ---------------------------------------------------------------------------
__global__ void ln_window_kernel(const float* __restrict__ src,
                                 const float* __restrict__ g,
                                 const float* __restrict__ bta,
                                 f16* __restrict__ dst, int S, int ntok) {
  int wid = threadIdx.x >> 5, lane = threadIdx.x & 31;
  int tok = blockIdx.x * 8 + wid;
  if (tok >= ntok) return;
  const float* p = src + (size_t)tok * DIMC;
  float v[8];
#pragma unroll
  for (int i = 0; i < 8; ++i) v[i] = p[lane * 8 + i];
  float s = 0.f;
#pragma unroll
  for (int i = 0; i < 8; ++i) s += v[i];
  s = warp_sum(s);
  float mu = s * (1.0f / DIMC);
  float q = 0.f;
#pragma unroll
  for (int i = 0; i < 8; ++i) { float d = v[i] - mu; q += d * d; }
  q = warp_sum(q);
  float rs = rsqrtf(q * (1.0f / DIMC) + 1e-5f);
  int b = tok / (S * 4096);
  int rem = tok % (S * 4096);
  int sidx = rem / 4096;
  int hr = (rem & 4095) >> 6, wc = rem & 63;
  int b_ = b * NW + (hr >> 3) * 8 + (wc >> 3);
  int n  = (hr & 7) * 8 + (wc & 7);
  f16* d = dst + (((size_t)b_ * S + sidx) * NTOK + n) * DIMC;
#pragma unroll
  for (int i = 0; i < 8; ++i) {
    int c = lane * 8 + i;
    d[c] = (f16)((v[i] - mu) * rs * g[c] + bta[c]);
  }
}

// ---------------------------------------------------------------------------
// Generic WMMA GEMM: out(M,NOUT) = A(M,K) * W(NOUT,K)^T + bias.
// A and W are f16 (W pre-converted). Wave computes a 32x64 tile (2 A-frags
// reuse each B-frag -> 2 WMMA per B load). Block = 8 waves = 256 rows.
// NOUT/GRP are compile-time so epilogue addressing folds to base+imm offsets.
// AMAP 0: identity rows. AMAP 1: pool-q rows (r -> (r/64)*384 + r%64).
// EPI 0: f32 store          1: f32 add-into
//     2: f16 store          3: f16 gelu store
//     4: *SCALEQ, (row/GRP, h, row%GRP, d) f16      (q / pool-q)
//     5: (row/GRP, h, row%GRP, d) f16               (k layouts)
//     6: (row/GRP, h, d, row%GRP) f16               (v transposed)
//     7: split kv: cols<256 -> k layout(out), cols>=256 -> vT(out2), GRP=64
// ---------------------------------------------------------------------------
template <int KDIM, int NOUT, int GRP, int AMAP, int EPI>
__global__ void gemm16(const f16* __restrict__ A, const f16* __restrict__ Wh,
                       const float* __restrict__ bias, void* __restrict__ out,
                       void* __restrict__ out2, int Mrows) {
  int wid = threadIdx.x >> 5, lane = threadIdx.x & 31;
  int mtile = (blockIdx.x * 8 + wid) * 2;          // two 16-row subtiles
  if (mtile * 16 >= Mrows) return;
  int sel = lane >> 4, l16 = lane & 15;
  int r0 = mtile * 16 + l16;
  int r1 = r0 + 16;
  size_t arow0 = (AMAP == 1) ? ((size_t)(r0 >> 6) * 384 + (r0 & 63)) : (size_t)r0;
  size_t arow1 = (AMAP == 1) ? ((size_t)(r1 >> 6) * 384 + (r1 & 63)) : (size_t)r1;
  const f16* abase0 = A + arow0 * KDIM + sel * 8;
  const f16* abase1 = A + arow1 * KDIM + sel * 8;
  int colbase = blockIdx.y * 64;
  v8f acc[2][4] = {};
  for (int kc = 0; kc < KDIM / 32; ++kc) {
    if (kc + 1 < KDIM / 32) {           // cover the strided A row fetch latency
      __builtin_prefetch(abase0 + (kc + 1) * 32, 0, 3);
      __builtin_prefetch(abase1 + (kc + 1) * 32, 0, 3);
    }
    v16h a0 = ldfragA(abase0 + kc * 32);
    v16h a1 = ldfragA(abase1 + kc * 32);
#pragma unroll
    for (int nt = 0; nt < 4; ++nt) {
      int col = colbase + nt * 16 + l16;
      v16h b = ldfragB(Wh + (size_t)col * KDIM + kc * 32 + sel * 8);
      acc[0][nt] = wmma_f16(a0, b, acc[0][nt]);
      acc[1][nt] = wmma_f16(a1, b, acc[1][nt]);
    }
  }
#pragma unroll
  for (int mi = 0; mi < 2; ++mi) {
    int row0 = (mtile + mi) * 16 + sel * 8;        // 8 consecutive rows/lane
    int gb   = row0 / GRP;                          // constant-folded divide
    int rin0 = row0 % GRP;                          // rows stay within group
#pragma unroll
    for (int nt = 0; nt < 4; ++nt) {
      int col = colbase + nt * 16 + l16;
      float bv = bias[col];
      if constexpr (EPI == 0 || EPI == 1) {
        float* o = (float*)out + (size_t)row0 * NOUT + col;
#pragma unroll
        for (int vr = 0; vr < 8; ++vr) {
          float v = acc[mi][nt][vr] + bv;
          if constexpr (EPI == 0) o[vr * NOUT] = v;
          else                    o[vr * NOUT] = o[vr * NOUT] + v;
        }
      } else if constexpr (EPI == 2 || EPI == 3) {
        f16* o = (f16*)out + (size_t)row0 * NOUT + col;
#pragma unroll
        for (int vr = 0; vr < 8; ++vr) {
          float v = acc[mi][nt][vr] + bv;
          o[vr * NOUT] = (f16)((EPI == 3) ? gelu_exact(v) : v);
        }
      } else if constexpr (EPI == 4 || EPI == 5) {
        f16* o = (f16*)out + ((((size_t)gb) * NH + (col >> 5)) * GRP + rin0) * HD + (col & 31);
#pragma unroll
        for (int vr = 0; vr < 8; ++vr) {
          float v = acc[mi][nt][vr] + bv;
          if constexpr (EPI == 4) v *= SCALEQ;
          o[vr * HD] = (f16)v;
        }
      } else if constexpr (EPI == 6) {
        f16* o = (f16*)out + ((((size_t)gb) * NH + (col >> 5)) * HD + (col & 31)) * GRP + rin0;
#pragma unroll
        for (int vr = 0; vr < 8; ++vr) o[vr] = (f16)(acc[mi][nt][vr] + bv);
      } else if constexpr (EPI == 7) {
        if (col < DIMC) {
          f16* o = (f16*)out + ((((size_t)gb) * NH + (col >> 5)) * 64 + rin0) * HD + (col & 31);
#pragma unroll
          for (int vr = 0; vr < 8; ++vr) o[vr * HD] = (f16)(acc[mi][nt][vr] + bv);
        } else {
          int j = col - DIMC;
          f16* o = (f16*)out2 + ((((size_t)gb) * NH + (j >> 5)) * HD + (j & 31)) * 64 + rin0;
#pragma unroll
          for (int vr = 0; vr < 8; ++vr) o[vr] = (f16)(acc[mi][nt][vr] + bv);
        }
      }
    }
  }
}

// ---------------------------------------------------------------------------
// Windowed cross-attention. grid = B_*T*M blocks of 256 (8 waves = 8 heads).
// Per wave: S = q k^T + rpb (rpb preloaded to LDS and folded into the WMMA C
// operand), f32 softmax in LDS, O = P v (WMMA).
// Dynamic LDS: 8 waves * (16384 S + 8192 P + 1024 rpb) = 204800 B.
// Output written flat as (B_, M, T, h, N, d) so the later flat read as
// (B_, T, M, N, C) reproduces the reference's transpose+reshape exactly.
// ---------------------------------------------------------------------------
__global__ void win_attn_kernel(const f16* __restrict__ q, const f16* __restrict__ k,
                                const f16* __restrict__ vt, const float* __restrict__ rpb,
                                f16* __restrict__ outp) {
  extern __shared__ char smem[];
  int wid = threadIdx.x >> 5, lane = threadIdx.x & 31;
  int h = wid;
  int bid = blockIdx.x;                    // B_*T*M = 2560
  int b_ = bid / (TT * MM);
  int rem = bid % (TT * MM);
  int t = rem / MM, m = rem % MM;
  int bt = b_ * TT + t, bm = b_ * MM + m;
  const f16* qh = q + ((size_t)(bt * NH + h)) * NTOK * HD;
  const f16* kh = k + ((size_t)(bm * NH + h)) * NTOK * HD;
  const f16* vh = vt + ((size_t)(bm * NH + h)) * HD * NTOK;
  float* S    = (float*)(smem + wid * 25600);
  f16*   P    = (f16*)(smem + wid * 25600 + 16384);
  float* rpbw = (float*)(smem + wid * 25600 + 24576);
  int sel = lane >> 4, l16 = lane & 15;

  // preload this head's 225-entry rpb column into LDS (one pass, coalesced)
  for (int i = lane; i < 225; i += 32) rpbw[i] = rpb[i * NH + h];
  __syncthreads();

  // S = (q*scale) k^T + rpb   (rpb enters through the WMMA C operand)
  for (int mt = 0; mt < 4; ++mt) {
    v16h a = ldfragA(qh + (mt * 16 + l16) * HD + sel * 8);
#pragma unroll
    for (int nt = 0; nt < 4; ++nt) {
      v16h b = ldfragB(kh + (nt * 16 + l16) * HD + sel * 8);
      v8f acc;
      int kc = nt * 16 + l16;
#pragma unroll
      for (int vr = 0; vr < 8; ++vr) {
        int n = mt * 16 + vr + sel * 8;
        int idx = ((n >> 3) - (kc >> 3) + 7) * 15 + ((n & 7) - (kc & 7) + 7);
        acc[vr] = rpbw[idx];
      }
      acc = wmma_f16(a, b, acc);
#pragma unroll
      for (int vr = 0; vr < 8; ++vr) {
        int n = mt * 16 + vr + sel * 8;
        S[n * 64 + kc] = acc[vr];
      }
    }
  }
  __syncthreads();
  // softmax: each lane owns 2 rows
#pragma unroll
  for (int rr = 0; rr < 2; ++rr) {
    int row = lane * 2 + rr;
    float* sr = S + row * 64;
    float mx = -1e30f;
    for (int i = 0; i < 64; ++i) mx = fmaxf(mx, sr[i]);
    float sum = 0.f;
    for (int i = 0; i < 64; ++i) { float e = __expf(sr[i] - mx); sr[i] = e; sum += e; }
    float inv = 1.0f / sum;
    f16* pr = P + row * 64;
    for (int i = 0; i < 64; ++i) pr[i] = (f16)(sr[i] * inv);
  }
  __syncthreads();
  // O = P @ V   (K = 64, two chunks)
  f16* ob = outp + (((size_t)(bm * TT + t)) * NH + h) * NTOK * HD;
  for (int mt = 0; mt < 4; ++mt) {
    v16h a0 = ldfragA((const f16*)P + (mt * 16 + l16) * 64 + 0 * 32 + sel * 8);
    v16h a1 = ldfragA((const f16*)P + (mt * 16 + l16) * 64 + 1 * 32 + sel * 8);
#pragma unroll
    for (int nt = 0; nt < 2; ++nt) {
      v16h b0 = ldfragB(vh + (nt * 16 + l16) * 64 + 0 * 32 + sel * 8);
      v16h b1 = ldfragB(vh + (nt * 16 + l16) * 64 + 1 * 32 + sel * 8);
      v8f acc = {};
      acc = wmma_f16(a0, b0, acc);
      acc = wmma_f16(a1, b1, acc);
#pragma unroll
      for (int vr = 0; vr < 8; ++vr) {
        int n = mt * 16 + vr + sel * 8;
        int d = nt * 16 + l16;
        ob[n * HD + d] = (f16)acc[vr];
      }
    }
  }
}

// ---------------------------------------------------------------------------
// Build pool tokens: tok[b2, slot, n, c] = (slot==0 ? mean_m xw : xw[slot-1])
//                                          + pos_emb[n,c], as f16.
// xw read flat from attention output (reshape identity).
// ---------------------------------------------------------------------------
__global__ void pool_prep_kernel(const f16* __restrict__ att, const float* __restrict__ pos,
                                 f16* __restrict__ tok) {
  int c = threadIdx.x;
  int bid = blockIdx.x;                 // B2 * 6 * 64
  int b2 = bid / 384;
  int rem = bid % 384;
  int slot = rem >> 6, n = rem & 63;
  const f16* base = att + (size_t)b2 * (MM * NTOK * DIMC) + n * DIMC + c;
  float val;
  if (slot == 0) {
    float s = 0.f;
#pragma unroll
    for (int m = 0; m < MM; ++m) s += (float)base[m * (NTOK * DIMC)];
    val = s * 0.2f;
  } else {
    val = (float)base[(slot - 1) * (NTOK * DIMC)];
  }
  val += pos[n * DIMC + c];
  tok[((size_t)b2 * 384 + rem) * DIMC + c] = (f16)val;
}

// ---------------------------------------------------------------------------
// Attention pooling core: one wave per (b2, head). q 64x32, k/v len 384.
// Dynamic LDS: 64*384 f32 (S) + 64*384 f16 (P) = 147456 B.
// ---------------------------------------------------------------------------
__global__ void pool_attn_kernel(const f16* __restrict__ q, const f16* __restrict__ k,
                                 const f16* __restrict__ vt, f16* __restrict__ outp) {
  extern __shared__ char smem[];
  int lane = threadIdx.x & 31;
  int b2 = blockIdx.x >> 3, h = blockIdx.x & 7;
  const f16* qh = q + ((size_t)(b2 * NH + h)) * 64 * HD;
  const f16* kh = k + ((size_t)(b2 * NH + h)) * 384 * HD;
  const f16* vh = vt + ((size_t)(b2 * NH + h)) * HD * 384;
  float* S = (float*)smem;
  f16*   P = (f16*)(smem + 64 * 384 * 4);
  int sel = lane >> 4, l16 = lane & 15;

  for (int mt = 0; mt < 4; ++mt) {
    v16h a = ldfragA(qh + (mt * 16 + l16) * HD + sel * 8);
    for (int nt = 0; nt < 24; ++nt) {
      v16h b = ldfragB(kh + (nt * 16 + l16) * HD + sel * 8);
      v8f acc = {};
      acc = wmma_f16(a, b, acc);
#pragma unroll
      for (int vr = 0; vr < 8; ++vr) {
        int n = mt * 16 + vr + sel * 8;
        int kc = nt * 16 + l16;
        S[n * 384 + kc] = acc[vr];
      }
    }
  }
  __syncthreads();
#pragma unroll
  for (int rr = 0; rr < 2; ++rr) {
    int row = lane * 2 + rr;
    float* sr = S + row * 384;
    float mx = -1e30f;
    for (int i = 0; i < 384; ++i) mx = fmaxf(mx, sr[i]);
    float sum = 0.f;
    for (int i = 0; i < 384; ++i) { float e = __expf(sr[i] - mx); sr[i] = e; sum += e; }
    float inv = 1.0f / sum;
    f16* pr = P + row * 384;
    for (int i = 0; i < 384; ++i) pr[i] = (f16)(sr[i] * inv);
  }
  __syncthreads();
  f16* ob = outp + (size_t)b2 * 64 * DIMC;
  for (int mt = 0; mt < 4; ++mt) {
#pragma unroll
    for (int nt = 0; nt < 2; ++nt) {
      v8f acc = {};
      for (int kc = 0; kc < 12; ++kc) {
        v16h a = ldfragA((const f16*)P + (mt * 16 + l16) * 384 + kc * 32 + sel * 8);
        v16h b = ldfragB(vh + (nt * 16 + l16) * 384 + kc * 32 + sel * 8);
        acc = wmma_f16(a, b, acc);
      }
#pragma unroll
      for (int vr = 0; vr < 8; ++vr) {
        int n = mt * 16 + vr + sel * 8;
        int d = nt * 16 + l16;
        ob[n * DIMC + h * HD + d] = (f16)acc[vr];
      }
    }
  }
}

// ---------------------------------------------------------------------------
// x1 = x + window_reverse(proj); write x1 to d_out; LN2(x1) -> f16 for MLP.
// ---------------------------------------------------------------------------
__global__ void resid_ln2_kernel(const float* __restrict__ x, const float* __restrict__ proj,
                                 const float* __restrict__ g, const float* __restrict__ bb,
                                 float* __restrict__ xout, f16* __restrict__ mlpin) {
  int wid = threadIdx.x >> 5, lane = threadIdx.x & 31;
  int tok = blockIdx.x * 8 + wid;                // 32768 tokens (B,T,H,W)
  int b = tok >> 14;
  int rem = tok & 16383;
  int t = rem >> 12;
  int hr = (rem & 4095) >> 6, wc = rem & 63;
  int b_ = b * NW + (hr >> 3) * 8 + (wc >> 3);
  int n  = (hr & 7) * 8 + (wc & 7);
  const float* pr = proj + (((size_t)(b_ * TT + t)) * NTOK + n) * DIMC;
  const float* xp = x + (size_t)tok * DIMC;
  float v[8];
#pragma unroll
  for (int i = 0; i < 8; ++i) {
    int c = lane * 8 + i;
    v[i] = xp[c] + pr[c];
  }
  float* xo = xout + (size_t)tok * DIMC;
#pragma unroll
  for (int i = 0; i < 8; ++i) xo[lane * 8 + i] = v[i];
  float s = 0.f;
#pragma unroll
  for (int i = 0; i < 8; ++i) s += v[i];
  s = warp_sum(s);
  float mu = s * (1.0f / DIMC);
  float qv = 0.f;
#pragma unroll
  for (int i = 0; i < 8; ++i) { float d = v[i] - mu; qv += d * d; }
  qv = warp_sum(qv);
  float rs = rsqrtf(qv * (1.0f / DIMC) + 1e-5f);
  f16* mo = mlpin + (size_t)tok * DIMC;
#pragma unroll
  for (int i = 0; i < 8; ++i) {
    int c = lane * 8 + i;
    mo[c] = (f16)((v[i] - mu) * rs * g[c] + bb[c]);
  }
}

// ---------------------------------------------------------------------------
extern "C" void kernel_launch(void* const* d_in, const int* in_sizes, int n_in,
                              void* d_out, int out_size, void* d_ws, size_t ws_size,
                              hipStream_t stream) {
  const float* x      = (const float*)d_in[0];
  const float* kv     = (const float*)d_in[1];
  const float* nqg    = (const float*)d_in[3];
  const float* nqb    = (const float*)d_in[4];
  const float* nkg    = (const float*)d_in[5];
  const float* nkb    = (const float*)d_in[6];
  const float* q_w    = (const float*)d_in[7];
  const float* q_b    = (const float*)d_in[8];
  const float* kv_w   = (const float*)d_in[9];
  const float* kv_b   = (const float*)d_in[10];
  const float* rpb    = (const float*)d_in[11];
  const float* pos    = (const float*)d_in[12];
  const float* pq_w   = (const float*)d_in[13];
  const float* pq_b   = (const float*)d_in[14];
  const float* pk_w   = (const float*)d_in[15];
  const float* pk_b   = (const float*)d_in[16];
  const float* pv_w   = (const float*)d_in[17];
  const float* pv_b   = (const float*)d_in[18];
  const float* po_w   = (const float*)d_in[19];
  const float* po_b   = (const float*)d_in[20];
  const float* proj_w = (const float*)d_in[21];
  const float* proj_b = (const float*)d_in[22];
  const float* n2g    = (const float*)d_in[23];
  const float* n2b    = (const float*)d_in[24];
  const float* w1     = (const float*)d_in[25];
  const float* b1     = (const float*)d_in[26];
  const float* w2     = (const float*)d_in[27];
  const float* b2v    = (const float*)d_in[28];

  char* ws = (char*)d_ws;
  const size_t MB = 1024ull * 1024ull;
  // Region plan (lifetime-based aliasing):
  f16* xkv16  = (f16*)(ws + 0);          // 20.97 MB   (dead after kv GEMM)
  f16* tok16  = (f16*)(ws + 0);          // 100.7 MB   (dead after pool GEMMs)
  float* projout = (float*)(ws + 0);     // 33.6 MB    (after tok16 dead)
  f16* xq16   = (f16*)(ws + 101 * MB);   // 16.8 MB    (dead after q GEMM)
  f16* k16p   = (f16*)(ws + 101 * MB);   // 100.7 MB
  f16* att16  = (f16*)(ws + 202 * MB);   // 83.9 MB    (dead after pool_prep)
  f16* vt16p  = (f16*)(ws + 202 * MB);   // 100.7 MB
  f16* qbuf   = (f16*)(ws + 303 * MB);   // 16.8 MB    (dead after win_attn)
  f16* q16p   = (f16*)(ws + 303 * MB);   // 16.8 MB    (dead after pool_attn)
  f16* mlpin  = (f16*)(ws + 303 * MB);   // 16.8 MB
  f16* k16    = (f16*)(ws + 320 * MB);   // 21.0 MB    (dead after win_attn)
  f16* poolo16= (f16*)(ws + 320 * MB);   // 16.8 MB
  f16* vT16   = (f16*)(ws + 341 * MB);   // 21.0 MB    (dead after win_attn)
  f16* gelu16 = (f16*)(ws + 341 * MB);   // 16.8 MB
  f16* h16    = (f16*)(ws + 362 * MB);   // 67.1 MB
  // f16 weight cache (persistent, ~2 MB; L2-resident)
  char* wb    = ws + 430 * MB;
  f16* qw16   = (f16*)(wb + 0);          //  65536 el
  f16* kvw16  = (f16*)(wb + 131072);     // 131072 el
  f16* pqw16  = (f16*)(wb + 393216);     //  65536 el
  f16* pkw16  = (f16*)(wb + 524288);
  f16* pvw16  = (f16*)(wb + 655360);
  f16* pow16  = (f16*)(wb + 786432);
  f16* prw16  = (f16*)(wb + 917504);
  f16* w1_16  = (f16*)(wb + 1048576);    // 262144 el
  f16* w2_16  = (f16*)(wb + 1572864);    // 262144 el

  // 0) convert all GEMM weights to f16 once
  cvt_w_kernel<<<64, 256, 0, stream>>>(q_w, qw16, 65536);
  cvt_w_kernel<<<128, 256, 0, stream>>>(kv_w, kvw16, 131072);
  cvt_w_kernel<<<64, 256, 0, stream>>>(pq_w, pqw16, 65536);
  cvt_w_kernel<<<64, 256, 0, stream>>>(pk_w, pkw16, 65536);
  cvt_w_kernel<<<64, 256, 0, stream>>>(pv_w, pvw16, 65536);
  cvt_w_kernel<<<64, 256, 0, stream>>>(po_w, pow16, 65536);
  cvt_w_kernel<<<64, 256, 0, stream>>>(proj_w, prw16, 65536);
  cvt_w_kernel<<<256, 256, 0, stream>>>(w1, w1_16, 262144);
  cvt_w_kernel<<<256, 256, 0, stream>>>(w2, w2_16, 262144);

  // 1-2) LayerNorm + window partition (f32 -> f16)
  ln_window_kernel<<<4096, 256, 0, stream>>>(x, nqg, nqb, xq16, TT, 32768);
  ln_window_kernel<<<5120, 256, 0, stream>>>(kv, nkg, nkb, xkv16, MM, 40960);

  // 3) q projection, *scale, split heads -> (bt,h,n,d)
  gemm16<256, 256, 64, 0, 4><<<dim3(128, 4), 256, 0, stream>>>(xq16, qw16, q_b, qbuf, nullptr, 32768);
  // 4) kv projection -> k (bm,h,n,d) + v transposed (bm,h,d,n)
  gemm16<256, 512, 64, 0, 7><<<dim3(160, 8), 256, 0, stream>>>(xkv16, kvw16, kv_b, k16, vT16, 40960);

  // 5) windowed attention (+rpb, softmax) -> att16 flat (B_,M,T,h,N,d)
  win_attn_kernel<<<2560, 256, 204800, stream>>>(qbuf, k16, vT16, rpb, att16);

  // 6) pool tokens: [avg | xw] + pos_emb
  pool_prep_kernel<<<196608, 256, 0, stream>>>(att16, pos, tok16);

  // 7-9) pool q/k/v projections
  gemm16<256, 256, 64, 1, 4><<<dim3(128, 4), 256, 0, stream>>>(tok16, pqw16, pq_b, q16p, nullptr, 32768);
  gemm16<256, 256, 384, 0, 5><<<dim3(768, 4), 256, 0, stream>>>(tok16, pkw16, pk_b, k16p, nullptr, 196608);
  gemm16<256, 256, 384, 0, 6><<<dim3(768, 4), 256, 0, stream>>>(tok16, pvw16, pv_b, vt16p, nullptr, 196608);

  // 10) pooling attention core
  pool_attn_kernel<<<4096, 32, 147456, stream>>>(q16p, k16p, vt16p, poolo16);

  // 11) output projection of pool, then GELU
  gemm16<256, 256, 64, 0, 3><<<dim3(128, 4), 256, 0, stream>>>(poolo16, pow16, po_b, gelu16, nullptr, 32768);
  // 12) proj
  gemm16<256, 256, 64, 0, 0><<<dim3(128, 4), 256, 0, stream>>>(gelu16, prw16, proj_b, projout, nullptr, 32768);

  // 13) residual + window reverse + LN2
  resid_ln2_kernel<<<4096, 256, 0, stream>>>(x, projout, n2g, n2b, (float*)d_out, mlpin);

  // 14) MLP fc1 + GELU
  gemm16<256, 1024, 64, 0, 3><<<dim3(128, 16), 256, 0, stream>>>(mlpin, w1_16, b1, h16, nullptr, 32768);
  // 15) MLP fc2, accumulate into x1 already in d_out
  gemm16<1024, 256, 64, 0, 1><<<dim3(128, 4), 256, 0, stream>>>(h16, w2_16, b2v, (float*)d_out, nullptr, 32768);
}
